// LinearAttentionLayer_62302795596459
// MI455X (gfx1250) — matrix operand order repaired and checked by reading, not compiled, not run on hardware
//
#include <hip/hip_runtime.h>

// ---------------------------------------------------------------------------
// Linear attention layer (chunked scan) for MI455X / gfx1250.
//  - all GEMMs on v_wmma_f32_16x16x32_f16 (wave32 WMMA, f32 accumulate)
//  - tile staging via GLOBAL_LOAD_ASYNC_TO_LDS_B128 (ASYNCcnt) when available
//  - scan K-chunk staged by the Tensor Data Mover (TENSOR_LOAD_TO_LDS)
// B=4 T=2048 D=1024 H=8 DK=64, chunk C=64.
// ---------------------------------------------------------------------------

typedef __attribute__((ext_vector_type(16))) _Float16 v16h;
typedef __attribute__((ext_vector_type(8)))  float    v8f;
typedef __attribute__((ext_vector_type(4)))  unsigned int v4u;
typedef __attribute__((ext_vector_type(8)))  int          v8i;
typedef __attribute__((ext_vector_type(4)))  int          v4i;

// Exact pointee type of the async-load builtin params: GCC-style <4 x i32>,
// in target address spaces 1 (global) and 3 (LDS).
typedef int v4i_vs __attribute__((vector_size(16)));
typedef __attribute__((address_space(1))) v4i_vs as1_v4i;
typedef __attribute__((address_space(3))) v4i_vs as3_v4i;

#define BB   4
#define TT   2048
#define DD   1024
#define HH   8
#define DKK  64
#define HD   512          // H*DK
#define MTOT 8192         // B*T
#define EPSV 1e-6f

// ---- feature probes -------------------------------------------------------
#if defined(__has_builtin)
#  if __has_builtin(__builtin_amdgcn_global_load_async_to_lds_b128) && \
      __has_builtin(__builtin_amdgcn_s_wait_asynccnt)
#    define HAVE_ASYNC 1
#  endif
#  if __has_builtin(__builtin_amdgcn_tensor_load_to_lds) && \
      __has_builtin(__builtin_amdgcn_s_wait_tensorcnt)
#    define HAVE_TDM 1
#  endif
#endif
#ifndef HAVE_ASYNC
#define HAVE_ASYNC 0
#endif
#ifndef HAVE_TDM
#define HAVE_TDM 0
#endif

// 16-byte global -> LDS copy. Async DMA path writes LDS without touching VGPRs.
// AS1 pointer built from the full 64-bit VA; AS3 pointer from the low 32 bits
// of the generic address (flat-aperture rule: LDS offset is the low 32 bits).
static __device__ __forceinline__ void g2l16(const _Float16* g, _Float16* l) {
#if HAVE_ASYNC
  __builtin_amdgcn_global_load_async_to_lds_b128(
      (as1_v4i*)(unsigned long long)(size_t)(const void*)g,
      (as3_v4i*)(unsigned int)(size_t)(void*)l, 0, 0);
#else
  *(uint4*)l = *(const uint4*)g;
#endif
}

static __device__ __forceinline__ void wait_async_all() {
#if HAVE_ASYNC
  __builtin_amdgcn_s_wait_asynccnt(0);
#endif
}

static __device__ __forceinline__ v8f wmma_f16(v16h a, v16h b, v8f c) {
  // (neg_a, A, neg_b, B, c_mod, C, reuse_a, reuse_b)
  return __builtin_amdgcn_wmma_f32_16x16x32_f16(false, a, false, b, (short)0, c,
                                                false, false);
}

// A operand: 16x32 f16 tile, row-major at p (row stride = stride), ISA 7.12.2.
static __device__ __forceinline__ v16h ldA(const _Float16* p, int stride) {
  const int lane = threadIdx.x & 31;
  const int m    = lane & 15;
  const int kh   = lane >> 4;
  const _Float16* rp = p + m * stride;
  v16h a;
#pragma unroll
  for (int g = 0; g < 8; ++g) {
    const int kb = kh * 16 + 2 * g;
    a[2 * g]     = rp[kb];
    a[2 * g + 1] = rp[kb + 1];
  }
  return a;
}

// B operand: 32x16 from a [K][N] row-major tile.
static __device__ __forceinline__ v16h ldB_kn(const _Float16* p, int stride) {
  const int lane = threadIdx.x & 31;
  const int n    = lane & 15;
  const int kh   = lane >> 4;
  v16h b;
#pragma unroll
  for (int g = 0; g < 8; ++g) {
    const int k  = kh * 16 + 2 * g;
    b[2 * g]     = p[k * stride + n];
    b[2 * g + 1] = p[(k + 1) * stride + n];
  }
  return b;
}

// B operand: 32x16 from a [N][K] row-major tile (B transposed in LDS).
static __device__ __forceinline__ v16h ldB_nk(const _Float16* p, int stride) {
  const int lane = threadIdx.x & 31;
  const int n    = lane & 15;
  const int kh   = lane >> 4;
  const _Float16* rp = p + n * stride + kh * 16;
  v16h b;
#pragma unroll
  for (int g = 0; g < 8; ++g) {
    b[2 * g]     = rp[2 * g];
    b[2 * g + 1] = rp[2 * g + 1];
  }
  return b;
}

// ---------------------------------------------------------------------------
// Kernel 0: f32 -> f16 conversion (one-time; amortized over all tile reuse).
// ---------------------------------------------------------------------------
__global__ __launch_bounds__(256) void la_cvt_f16(const float* __restrict__ s,
                                                  _Float16* __restrict__ d,
                                                  int n) {
  int i = blockIdx.x * 2048 + threadIdx.x;
#pragma unroll
  for (int u = 0; u < 8; ++u, i += 256)
    if (i < n) d[i] = (_Float16)s[i];
}

// ---------------------------------------------------------------------------
// Kernel 1: fused projections, all-f16 operands. Out = act(X @ W^T).
//   which = blockIdx.y>>3 : 0->Q (elu+1), 1->K (elu+1), 2->V (id), 3->gate.
// 256 threads (8 waves), 64x64 tile, k-step 32, async-staged LDS tiles.
// ---------------------------------------------------------------------------
__global__ __launch_bounds__(256) void la_proj_wmma(
    const _Float16* __restrict__ x16,
    const _Float16* __restrict__ Wq, const _Float16* __restrict__ Wk,
    const _Float16* __restrict__ Wv, const _Float16* __restrict__ Wg,
    const float* __restrict__ gb,
    _Float16* __restrict__ Qo, _Float16* __restrict__ Ko,
    _Float16* __restrict__ Vo, _Float16* __restrict__ Go) {
  __shared__ __align__(16) _Float16 As[64][32];
  __shared__ __align__(16) _Float16 Bs[64][32];

  const int m0    = blockIdx.x * 64;
  const int which = blockIdx.y >> 3;
  const int n0    = (blockIdx.y & 7) * 64;
  const _Float16* __restrict__ W =
      (which == 0) ? Wq : (which == 1) ? Wk : (which == 2) ? Wv : Wg;
  _Float16* __restrict__ Out =
      (which == 0) ? Qo : (which == 1) ? Ko : (which == 2) ? Vo : Go;

  const int tid   = threadIdx.x;
  const int wave  = tid >> 5;
  const int tile0 = wave * 2;     // two adjacent tiles share the A operand
  const int mi    = tile0 >> 2;
  const int ni0   = tile0 & 3;
  const int sr    = tid >> 2;            // staging row 0..63
  const int sc    = (tid & 3) * 8;       // staging col (f16 units)

  v8f acc0 = {};
  v8f acc1 = {};

  for (int k0 = 0; k0 < DD; k0 += 32) {
    g2l16(&x16[(size_t)(m0 + sr) * DD + k0 + sc], &As[sr][sc]);
    g2l16(&W[(size_t)(n0 + sr) * DD + k0 + sc], &Bs[sr][sc]);
    wait_async_all();
    __syncthreads();
    v16h a  = ldA(&As[mi * 16][0], 32);
    v16h b0 = ldB_nk(&Bs[ni0 * 16][0], 32);
    v16h b1 = ldB_nk(&Bs[(ni0 + 1) * 16][0], 32);
    acc0 = wmma_f16(a, b0, acc0);
    acc1 = wmma_f16(a, b1, acc1);
    __syncthreads();
  }

  const int lane = tid & 31;
  const int n = lane & 15, mh = lane >> 4;
#pragma unroll
  for (int u = 0; u < 2; ++u) {
    v8f acc = u ? acc1 : acc0;
    const int ni = ni0 + u;
#pragma unroll
    for (int r = 0; r < 8; ++r) {
      const int m  = m0 + mi * 16 + mh * 8 + r;
      const int nn = n0 + ni * 16 + n;
      float v = acc[r];
      if (which <= 1) {
        v = (v > 0.f) ? (v + 1.f) : __expf(v);          // elu(v)+1
      } else if (which == 3) {
        v = 1.f / (1.f + __expf(-(v + gb[nn])));        // sigmoid(v + b)
      }
      Out[(size_t)m * HD + nn] = (_Float16)v;
    }
  }
}

// ---------------------------------------------------------------------------
// Kernel 2: chunked decayed linear-attention scan. One block per (b,h),
// 128 threads (4 waves). K-chunk staged by the Tensor Data Mover; Q/V by
// async copies. Per 64-step chunk:
//   Raw = Q K^T ; A = mask*gamma^(t-j)*Raw       (f32 epilogue)
//   O   = gamma^(t+1)*(Q @ S) + A @ V ; den = gamma^(t+1) q.z + rowsum(A)+eps
//   S  <- gamma^64 * S + (gamma^(63-j) K)^T @ V  (WMMA accumulate operand)
// Writes gated output (O * gate) in f16 for the final GEMM.
// ---------------------------------------------------------------------------
__global__ __launch_bounds__(128) void la_scan_wmma(
    const _Float16* __restrict__ Qg, const _Float16* __restrict__ Kg,
    const _Float16* __restrict__ Vg, const _Float16* __restrict__ Gg,
    const float* __restrict__ dlog, _Float16* __restrict__ OGg) {
  const int b = blockIdx.x >> 3;
  const int h = blockIdx.x & 7;

  __shared__ __align__(16) _Float16 Qs[64 * 64];    // [t][d]
  __shared__ __align__(16) _Float16 Ks[64 * 64];    // [j][d]   (TDM target)
  __shared__ __align__(16) _Float16 KdT[64 * 64];   // [d][j] = gamma^(63-j)*K[j][d]
  __shared__ __align__(16) _Float16 Vs[64 * 64];    // [j][dv]
  __shared__ __align__(16) _Float16 Adec[64 * 64];  // [t][j]
  __shared__ __align__(16) _Float16 S16[64 * 64];   // state [d][dv]
  __shared__ float zv[64];
  __shared__ float den[64];
  __shared__ float gpow[65];

  const int tid  = threadIdx.x;
  const int wave = tid >> 5;
  const int lane = tid & 31;
  const int n    = lane & 15;
  const int mh   = lane >> 4;
  const int mi   = wave;              // each wave owns 16 output rows

  const float gamma = 1.f / (1.f + __expf(-dlog[h]));
  if (tid <= 64) gpow[tid] = __powf(gamma, (float)tid);
  if (tid < 64) zv[tid] = 0.f;
  for (int i = tid; i < 64 * 64; i += 128) S16[i] = (_Float16)0.f;
  __syncthreads();

  for (int c = 0; c < 32; ++c) {
    const int t0 = c * 64;
    const size_t base = (size_t)(b * TT + t0) * HD + h * DKK;
    // ---- stage chunk: K via TDM, Q/V via async copies ----
#if HAVE_TDM
    if (wave == 0) {
      const unsigned long long ga = (unsigned long long)(size_t)(const void*)(Kg + base);
      const unsigned int ldsK = (unsigned int)(size_t)(void*)&Ks[0];
      // D# group0: count=1 | lds_addr | global_addr | type=2  (ISA 8.3)
      v4u g0 = {1u, ldsK, (unsigned int)ga,
                ((unsigned int)(ga >> 32) & 0x01FFFFFFu) | 0x80000000u};
      // D# group1: data_size=2B; tensor_dim0=64, tensor_dim1=64;
      // tile_dim0=64, tile_dim1=64; tensor_dim0_stride=512  (ISA 8.4)
      v8i g1 = {(int)(1u << 16), (int)(64u << 16), (int)(64u << 16),
                (int)(64u << 16), 64, 512, 0, 0};
      v4i gz = {0, 0, 0, 0};
#if defined(__clang_major__) && __clang_major__ >= 23
      v8i gz8 = {0, 0, 0, 0, 0, 0, 0, 0};
      __builtin_amdgcn_tensor_load_to_lds(g0, g1, gz, gz, gz8, 0);
#else
      __builtin_amdgcn_tensor_load_to_lds(g0, g1, gz, gz, 0);
#endif
      __builtin_amdgcn_s_wait_tensorcnt(0);
    }
#else
    for (int u = tid; u < 512; u += 128)
      g2l16(Kg + base + (size_t)(u >> 3) * HD + (u & 7) * 8,
            &Ks[(u >> 3) * 64 + (u & 7) * 8]);
#endif
    for (int u = tid; u < 512; u += 128) {
      const size_t go = base + (size_t)(u >> 3) * HD + (u & 7) * 8;
      const int lo = (u >> 3) * 64 + (u & 7) * 8;
      g2l16(Qg + go, &Qs[lo]);
      g2l16(Vg + go, &Vs[lo]);
    }
    wait_async_all();
    __syncthreads();
    // ---- KdT (decayed, transposed K) + den init ----
    for (int i = tid; i < 64 * 64; i += 128) {
      const int r = i & 63, d = i >> 6;
      KdT[d * 64 + r] = (_Float16)((float)Ks[r * 64 + d] * gpow[63 - r]);
    }
    if (tid < 64) {
      float s = 0.f;
      for (int d = 0; d < 64; ++d) s += (float)Qs[tid * 64 + d] * zv[d];
      den[tid] = s * gpow[tid + 1] + EPSV;
    }
    __syncthreads();
    // ---- matmul 1: Raw = Q K^T ; decay+mask -> Adec ; rowsum -> den ----
    {
      v16h a0 = ldA(&Qs[(mi * 16) * 64 + 0], 64);
      v16h a1 = ldA(&Qs[(mi * 16) * 64 + 32], 64);
#pragma unroll
      for (int ni = 0; ni < 4; ++ni) {
        v8f acc = {};
        v16h b0 = ldB_nk(&Ks[(ni * 16) * 64 + 0], 64);
        v16h b1 = ldB_nk(&Ks[(ni * 16) * 64 + 32], 64);
        acc = wmma_f16(a0, b0, acc);
        acc = wmma_f16(a1, b1, acc);
#pragma unroll
        for (int r = 0; r < 8; ++r) {
          const int t = mi * 16 + mh * 8 + r;
          const int j = ni * 16 + n;
          const float v = (t >= j) ? acc[r] * gpow[t - j] : 0.f;
          Adec[t * 64 + j] = (_Float16)v;
          atomicAdd(&den[t], v);          // ds_add_f32
        }
      }
    }
    __syncthreads();
    // ---- matmul 2+3: O = gamma^(t+1)*(Q @ S16) + Adec @ V ; gate; store ----
    {
      v16h a0 = ldA(&Qs[(mi * 16) * 64 + 0], 64);
      v16h a1 = ldA(&Qs[(mi * 16) * 64 + 32], 64);
      v16h e0 = ldA(&Adec[(mi * 16) * 64 + 0], 64);
      v16h e1 = ldA(&Adec[(mi * 16) * 64 + 32], 64);
#pragma unroll
      for (int ni = 0; ni < 4; ++ni) {
        v8f inter = {};
        v16h s0 = ldB_kn(&S16[0 * 64 + ni * 16], 64);
        v16h s1 = ldB_kn(&S16[32 * 64 + ni * 16], 64);
        inter = wmma_f16(a0, s0, inter);
        inter = wmma_f16(a1, s1, inter);
        v8f intra = {};
        v16h v0 = ldB_kn(&Vs[0 * 64 + ni * 16], 64);
        v16h v1 = ldB_kn(&Vs[32 * 64 + ni * 16], 64);
        intra = wmma_f16(e0, v0, intra);
        intra = wmma_f16(e1, v1, intra);
#pragma unroll
        for (int r = 0; r < 8; ++r) {
          const int t  = mi * 16 + mh * 8 + r;
          const int dv = ni * 16 + n;
          const size_t gi = (size_t)(b * TT + t0 + t) * HD + h * DKK + dv;
          const float o = (inter[r] * gpow[t + 1] + intra[r]) / den[t];
          OGg[gi] = (_Float16)(o * (float)Gg[gi]);
        }
      }
    }
    __syncthreads();
    // ---- matmul 4: S16 <- gamma^64 * S16 + KdT @ V (accumulate operand) ----
    {
      v16h a0 = ldA(&KdT[(mi * 16) * 64 + 0], 64);
      v16h a1 = ldA(&KdT[(mi * 16) * 64 + 32], 64);
#pragma unroll
      for (int ni = 0; ni < 4; ++ni) {
        v8f acc;
#pragma unroll
        for (int r = 0; r < 8; ++r)
          acc[r] = gpow[64] * (float)S16[(mi * 16 + mh * 8 + r) * 64 + ni * 16 + n];
        v16h v0 = ldB_kn(&Vs[0 * 64 + ni * 16], 64);
        v16h v1 = ldB_kn(&Vs[32 * 64 + ni * 16], 64);
        acc = wmma_f16(a0, v0, acc);
        acc = wmma_f16(a1, v1, acc);
#pragma unroll
        for (int r = 0; r < 8; ++r)
          S16[(mi * 16 + mh * 8 + r) * 64 + ni * 16 + n] = (_Float16)acc[r];
      }
    }
    __syncthreads();
    // ---- z update: z <- gamma^64 z + sum_j gamma^(63-j) k_j ----
    if (tid < 64) {
      float s = gpow[64] * zv[tid];
      for (int j = 0; j < 64; ++j) s += (float)KdT[tid * 64 + j];
      zv[tid] = s;
    }
    __syncthreads();
  }
}

// ---------------------------------------------------------------------------
// Kernel 3: y = (O*gate) @ W_O^T, all-f16 operands, f32 result.
// ---------------------------------------------------------------------------
__global__ __launch_bounds__(256) void la_out_wmma(
    const _Float16* __restrict__ OG, const _Float16* __restrict__ Wo,
    float* __restrict__ y) {
  __shared__ __align__(16) _Float16 As[64][32];
  __shared__ __align__(16) _Float16 Bs[64][32];

  const int m0 = blockIdx.x * 64;
  const int n0 = blockIdx.y * 64;
  const int tid   = threadIdx.x;
  const int wave  = tid >> 5;
  const int tile0 = wave * 2;
  const int mi    = tile0 >> 2;
  const int ni0   = tile0 & 3;
  const int sr    = tid >> 2;
  const int sc    = (tid & 3) * 8;

  v8f acc0 = {};
  v8f acc1 = {};

  for (int k0 = 0; k0 < HD; k0 += 32) {
    g2l16(&OG[(size_t)(m0 + sr) * HD + k0 + sc], &As[sr][sc]);
    g2l16(&Wo[(size_t)(n0 + sr) * HD + k0 + sc], &Bs[sr][sc]);
    wait_async_all();
    __syncthreads();
    v16h a  = ldA(&As[mi * 16][0], 32);
    v16h b0 = ldB_nk(&Bs[ni0 * 16][0], 32);
    v16h b1 = ldB_nk(&Bs[(ni0 + 1) * 16][0], 32);
    acc0 = wmma_f16(a, b0, acc0);
    acc1 = wmma_f16(a, b1, acc1);
    __syncthreads();
  }

  const int lane = tid & 31;
  const int n = lane & 15, mh = lane >> 4;
#pragma unroll
  for (int u = 0; u < 2; ++u) {
    v8f acc = u ? acc1 : acc0;
    const int ni = ni0 + u;
#pragma unroll
    for (int r = 0; r < 8; ++r) {
      const int m  = m0 + mi * 16 + mh * 8 + r;
      const int nn = n0 + ni * 16 + n;
      y[(size_t)m * DD + nn] = acc[r];
    }
  }
}

// ---------------------------------------------------------------------------
extern "C" void kernel_launch(void* const* d_in, const int* in_sizes, int n_in,
                              void* d_out, int out_size, void* d_ws, size_t ws_size,
                              hipStream_t stream) {
  const float* x    = (const float*)d_in[0];
  const float* Wq   = (const float*)d_in[1];
  const float* Wk   = (const float*)d_in[2];
  const float* Wv   = (const float*)d_in[3];
  const float* Wg   = (const float*)d_in[4];
  const float* gb   = (const float*)d_in[5];
  const float* Wo   = (const float*)d_in[6];
  const float* dlog = (const float*)d_in[7];
  float* y = (float*)d_out;

  // Workspace layout (f16): x16 | Wq16..Wg16 | Wo16 | Q|K|V|G | OG  (~61 MB)
  const size_t SX = (size_t)MTOT * DD;   // 8M
  const size_t SW = (size_t)HD * DD;     // 0.5M
  const size_t SP = (size_t)MTOT * HD;   // 4M
  _Float16* x16  = (_Float16*)d_ws;
  _Float16* wq16 = x16 + SX;
  _Float16* wk16 = wq16 + SW;
  _Float16* wv16 = wk16 + SW;
  _Float16* wg16 = wv16 + SW;
  _Float16* wo16 = wg16 + SW;
  _Float16* Q16  = wo16 + SW;
  _Float16* K16  = Q16 + SP;
  _Float16* V16  = K16 + SP;
  _Float16* G16  = V16 + SP;
  _Float16* OG16 = G16 + SP;

  la_cvt_f16<<<(int)((SX + 2047) / 2048), 256, 0, stream>>>(x, x16, (int)SX);
  la_cvt_f16<<<(int)((SW + 2047) / 2048), 256, 0, stream>>>(Wq, wq16, (int)SW);
  la_cvt_f16<<<(int)((SW + 2047) / 2048), 256, 0, stream>>>(Wk, wk16, (int)SW);
  la_cvt_f16<<<(int)((SW + 2047) / 2048), 256, 0, stream>>>(Wv, wv16, (int)SW);
  la_cvt_f16<<<(int)((SW + 2047) / 2048), 256, 0, stream>>>(Wg, wg16, (int)SW);
  la_cvt_f16<<<(int)((SW + 2047) / 2048), 256, 0, stream>>>(Wo, wo16, (int)SW);

  la_proj_wmma<<<dim3(MTOT / 64, 32), 256, 0, stream>>>(
      x16, wq16, wk16, wv16, wg16, gb, Q16, K16, V16, G16);
  la_scan_wmma<<<BB * HH, 128, 0, stream>>>(Q16, K16, V16, G16, dlog, OG16);
  la_out_wmma<<<dim3(MTOT / 64, DD / 64), 256, 0, stream>>>(OG16, wo16, y);
}
